// MetaUpSampler_22557168238771
// MI455X (gfx1250) — matrix operand order repaired and checked by reading, not compiled, not run on hardware
//
#include <hip/hip_runtime.h>

// Problem constants (from reference setup_inputs)
#define KC_N    4
#define KC_C    64
#define KC_H    192
#define KC_W    192
#define KC_OUTC 3
#define KC_S    4          // sr_scale
#define NTOT    48         // s*s*outC
#define KTOT    576        // C*9
#define KSTEPS  18         // KTOT/32

// Tiling
#define WTILE      64      // pixels per workgroup (one row segment)
#define SLAB_COLS  66      // WTILE + 2 halo
#define SPAT_STRIDE 144    // bytes per spatial point: 64ch*2B + 16B pad (16B aligned)

typedef __attribute__((ext_vector_type(16))) __bf16 v16bf;
typedef __attribute__((ext_vector_type(8)))  float  v8f;

union Frag { uint4 q[2]; v16bf v; };

__device__ __forceinline__ unsigned short f2bf(float x) {
  union { float f; unsigned u; } v; v.f = x;
  unsigned r = v.u + 0x7FFFu + ((v.u >> 16) & 1u);   // round-to-nearest-even
  return (unsigned short)(r >> 16);
}

// ---------------------------------------------------------------------------
// Prep kernel: Pos2Weight MLP -> B matrix (bf16) in WMMA B-fragment order.
// B element e = ((ks*3 + nt)*32 + lane)*16 + idx, where in the contiguous k order
// used here (same as A): lane<16 -> N=lane, K=idx (0..15); lane>=16 -> N=lane-16,
// K=16+idx. Permuted K index kp = ks*32 + kk maps to t = kp/64, c = kp%64,
// and the reference weight column is (c*9 + t)*3 + o with nn = nt*16+lane%16,
// ssub = nn/3, o = nn%3.
// ---------------------------------------------------------------------------
__global__ void prep_weights(const float* __restrict__ W1,
                             const float* __restrict__ b1,
                             const float* __restrict__ W2,
                             const float* __restrict__ b2,
                             unsigned short* __restrict__ Bws) {
  __shared__ float hidden[16][256];
  const int tid = threadIdx.x;   // 256 threads
  for (int s = 0; s < 16; ++s) {
    float p0 = 0.25f;
    float p1 = (float)(s >> 2) * 0.25f;
    float p2 = (float)(s & 3) * 0.25f;
    float h = p0 * W1[tid] + p1 * W1[256 + tid] + p2 * W1[512 + tid] + b1[tid];
    hidden[s][tid] = h > 0.f ? h : 0.f;
  }
  __syncthreads();

  const int e    = blockIdx.x * 256 + tid;   // < 18*3*32*16 = 27648
  const int idx  = e & 15;
  const int lane = (e >> 4) & 31;
  const int fr   = e >> 9;                   // ks*3 + nt
  const int nt   = fr % 3;
  const int ks   = fr / 3;

  const int kk   = ((lane >> 4) << 4) + idx; // (lane>=16)*16 + idx
  const int kp   = ks * 32 + kk;             // permuted K: t*64 + c
  const int t    = kp >> 6;
  const int c    = kp & 63;
  const int nn   = nt * 16 + (lane & 15);
  const int ssub = nn / 3;
  const int o    = nn - ssub * 3;
  const int col  = (c * 9 + t) * 3 + o;      // 0..1727

  float acc = b2[col];
  const float* h = hidden[ssub];
  for (int j = 0; j < 256; ++j)
    acc += h[j] * W2[j * 1728 + col];
  Bws[e] = f2bf(acc);
}

// ---------------------------------------------------------------------------
// Main kernel: stage 64ch x 3row x 66col bf16 slab (channel-minor) in LDS,
// then each wave does a 16(pixels) x 48(filters) x 576(K) WMMA GEMM, then
// transposes the result tile through LDS and drains it with async
// LDS->memory b128 stores (no VGPR bounce).
// ---------------------------------------------------------------------------
__global__ void __launch_bounds__(128) meta_upsample(
    const float* __restrict__ x,
    const unsigned short* __restrict__ Bws,
    float* __restrict__ out) {
  extern __shared__ char slab[];

  const int n   = blockIdx.z;
  const int gh  = blockIdx.y;
  const int w0  = blockIdx.x * WTILE;
  const int tid = threadIdx.x;
  const int lane = tid & 31;
  const int wave = tid >> 5;
  const int m16  = lane & 15;
  const int hsel = (lane >> 4) & 1;

  // ---- stage input slab: bf16, layout [row][col][channel], padded rows ----
  // Interior (cols w0..w0+63): vectorized float4 loads, 16B-aligned in global.
  for (int i = tid; i < KC_C * 3 * 16; i += 128) {
    int q  = i & 15;               // quad of columns
    int rr = i >> 4;               // c*3 + r
    int r  = rr % 3;
    int c  = rr / 3;
    int gr = gh + r - 1;
    float4 v = make_float4(0.f, 0.f, 0.f, 0.f);
    if ((unsigned)gr < (unsigned)KC_H)
      v = *(const float4*)&x[(((size_t)n * KC_C + c) * KC_H + gr) * KC_W + (w0 + q * 4)];
    char* p = slab + (size_t)(r * SLAB_COLS + 1 + q * 4) * SPAT_STRIDE + c * 2;
    *(unsigned short*)(p + 0 * SPAT_STRIDE) = f2bf(v.x);
    *(unsigned short*)(p + 1 * SPAT_STRIDE) = f2bf(v.y);
    *(unsigned short*)(p + 2 * SPAT_STRIDE) = f2bf(v.z);
    *(unsigned short*)(p + 3 * SPAT_STRIDE) = f2bf(v.w);
  }
  // Halo columns sc=0 (gc=w0-1) and sc=65 (gc=w0+64): scalar with bounds.
  for (int i = tid; i < KC_C * 3 * 2; i += 128) {
    int e  = i & 1;
    int rr = i >> 1;               // c*3 + r
    int r  = rr % 3;
    int c  = rr / 3;
    int gr = gh + r - 1;
    int sc = e ? (SLAB_COLS - 1) : 0;
    int gc = w0 + sc - 1;
    float v = 0.f;
    if ((unsigned)gr < (unsigned)KC_H && (unsigned)gc < (unsigned)KC_W)
      v = x[(((size_t)n * KC_C + c) * KC_H + gr) * KC_W + gc];
    *(unsigned short*)(slab + (size_t)(r * SLAB_COLS + sc) * SPAT_STRIDE + c * 2) = f2bf(v);
  }
  __syncthreads();

  v8f acc0 = {}, acc1 = {}, acc2 = {};
  const uint4* __restrict__ bq = (const uint4*)Bws;

  for (int ks = 0; ks < KSTEPS; ++ks) {
    const int t  = ks >> 1;
    const int cb = ks & 1;
    const int di = t / 3;
    const int dj = t - di * 3;

    // A fragment: 16 pixels x 32 channels at shift (di,dj), channel half cb.
    const int sc   = wave * 16 + m16 + dj;
    const int aoff = (di * SLAB_COLS + sc) * SPAT_STRIDE + (cb * 32 + hsel * 16) * 2;
    Frag a;
    a.q[0] = *(const uint4*)(slab + aoff);        // channels base .. base+7
    a.q[1] = *(const uint4*)(slab + aoff + 16);   // channels base+8 .. base+15

    // B fragments (fragment-ordered in workspace; 32B per lane, coalesced)
    const int bbase = (ks * 3) * 32 + lane;       // in 32B fragments-of-lane
    if (ks + 1 < KSTEPS)
      __builtin_prefetch(&bq[((ks + 1) * 3 * 32 + lane) * 2], 0, 1);
    Frag b0, b1, b2;
    b0.q[0] = bq[(bbase +   0) * 2 + 0];
    b0.q[1] = bq[(bbase +   0) * 2 + 1];
    b1.q[0] = bq[(bbase +  32) * 2 + 0];
    b1.q[1] = bq[(bbase +  32) * 2 + 1];
    b2.q[0] = bq[(bbase +  64) * 2 + 0];
    b2.q[1] = bq[(bbase +  64) * 2 + 1];

    acc0 = __builtin_amdgcn_wmma_f32_16x16x32_bf16(false, a.v, false, b0.v,
                                                   (short)0, acc0, false, false);
    acc1 = __builtin_amdgcn_wmma_f32_16x16x32_bf16(false, a.v, false, b1.v,
                                                   (short)0, acc1, false, false);
    acc2 = __builtin_amdgcn_wmma_f32_16x16x32_bf16(false, a.v, false, b2.v,
                                                   (short)0, acc2, false, false);
  }

  // ---- transpose result tile through LDS for coalesced stores ----
  // Output-row layout: seg = o*4+si (12 segments), each 256 consecutive
  // floats covering cols (w0..w0+63)*4 + sj for sj=0..3.
  __syncthreads();                 // slab reads done; reuse buffer
  float* outLds = (float*)slab;    // 12 * 256 * 4 = 12288 B  (< slab size)

#pragma unroll
  for (int nt = 0; nt < 3; ++nt) {
    const v8f acc = (nt == 0) ? acc0 : ((nt == 1) ? acc1 : acc2);
    const int nn   = nt * 16 + m16;
    const int ssub = nn / 3;
    const int o    = nn - ssub * 3;
    const int si   = ssub >> 2;
    const int sj   = ssub & 3;
    const int seg  = o * 4 + si;
#pragma unroll
    for (int v = 0; v < 8; ++v) {
      const int wl = wave * 16 + hsel * 8 + v;   // local pixel 0..63
      outLds[seg * 256 + wl * 4 + sj] = acc[v];
    }
  }
  __syncthreads();

  // 12 segments * 64 b128 = 768 async LDS->memory stores, coalesced per
  // segment. ASYNCcnt is drained by S_ENDPGM's implicit wait-idle, and each
  // wave's LDS stays live until its own counters hit zero.
  // (Dynamic LDS base is offset 0: this kernel has no static __shared__.)
  for (int i = tid; i < 12 * 64; i += 128) {
    const int seg = i >> 6;        // o*4 + si
    const int q   = i & 63;
    const int o   = seg >> 2;
    const int si  = seg & 3;
    const size_t base =
        (((size_t)n * KC_OUTC + o) * (KC_S * KC_H) + (gh * 4 + si)) * (size_t)(KC_S * KC_W)
        + (size_t)w0 * 4;
    unsigned long long ga = (unsigned long long)(uintptr_t)(out + base + (size_t)q * 4);
    unsigned loff = (unsigned)(seg * 64 + q) * 16u;
    asm volatile("global_store_async_from_lds_b128 %0, %1, off"
                 :: "v"(ga), "v"(loff)
                 : "memory");
  }
}

// ---------------------------------------------------------------------------
extern "C" void kernel_launch(void* const* d_in, const int* in_sizes, int n_in,
                              void* d_out, int out_size, void* d_ws, size_t ws_size,
                              hipStream_t stream) {
  const float* x  = (const float*)d_in[0];
  const float* W1 = (const float*)d_in[1];
  const float* b1 = (const float*)d_in[2];
  const float* W2 = (const float*)d_in[3];
  const float* b2 = (const float*)d_in[4];
  // d_in[5] = sr_scale (int scalar) — fixed to 4 in this problem size.

  unsigned short* Bws = (unsigned short*)d_ws;   // 27648 bf16 = 55296 B

  prep_weights<<<108, 256, 0, stream>>>(W1, b1, W2, b2, Bws);

  dim3 grid(KC_W / WTILE, KC_H, KC_N);
  const size_t shmem = (size_t)3 * SLAB_COLS * SPAT_STRIDE;  // 28512 B
  meta_upsample<<<grid, 128, shmem, stream>>>(x, Bws, (float*)d_out);
}
